// Attention_2740189134991
// MI455X (gfx1250) — matrix operand order repaired and checked by reading, not compiled, not run on hardware
//
#include <hip/hip_runtime.h>

// ---------------------------------------------------------------------------
// Types for CDNA5 WMMA (wave32): v_wmma_f32_16x16x32_bf16
// ---------------------------------------------------------------------------
typedef __bf16 bf16_t;
typedef __attribute__((ext_vector_type(16))) __bf16 v16bf;
typedef __attribute__((ext_vector_type(8)))  __bf16 v8bf;
typedef __attribute__((ext_vector_type(8)))  float  v8f;

#define SCALE_F 0.125f   // DH^-0.5, DH=64

// ---------------------------------------------------------------------------
// Fragment loaders per CDNA5 ISA 7.12.2 VGPR layouts (wave32).
//
// A (16x32 MxK, bf16): lane L holds row M = L&15.
//   lanes 0-15 : elems 0..7 -> K=k0+0..7,    elems 8..15 -> K=k0+16..23
//   lanes 16-31: elems 0..7 -> K=k0+8..15,   elems 8..15 -> K=k0+24..31
// src is row-major with leading dim ld (elements).
// ---------------------------------------------------------------------------
__device__ __forceinline__ v16bf load_a_frag(const bf16_t* __restrict__ src,
                                             int ld, int row0, int k0, int lane) {
    const int row = row0 + (lane & 15);
    const int kb  = (lane < 16) ? 0 : 8;
    const bf16_t* p = src + (size_t)row * ld + k0 + kb;
    union { v16bf v; v8bf h[2]; } u;
    u.h[0] = *(const v8bf*)(p);        // K = k0+kb   .. +7
    u.h[1] = *(const v8bf*)(p + 16);   // K = k0+kb+16.. +23
    return u.v;
}

// B (32x16 KxN, bf16): lane L holds column N = L&15.
//   lanes 0-15 : elems 0..7 -> K=k0+0..7,    elems 8..15 -> K=k0+16..23
//   lanes 16-31: elems 0..7 -> K=k0+8..15,   elems 8..15 -> K=k0+24..31
// src is "column-contiguous": element (kk, n) at src[(col0+n)*ld + k0+kk].
// ---------------------------------------------------------------------------
__device__ __forceinline__ v16bf load_b_frag(const bf16_t* __restrict__ src,
                                             int ld, int col0, int k0, int lane) {
    const int col = col0 + (lane & 15);
    const int kb  = (lane < 16) ? 0 : 8;
    const bf16_t* p = src + (size_t)col * ld + k0 + kb;
    union { v16bf v; v8bf h[2]; } u;
    u.h[0] = *(const v8bf*)(p);
    u.h[1] = *(const v8bf*)(p + 16);
    return u.v;
}

__device__ __forceinline__ v8f wmma_bf16(v16bf a, v16bf b, v8f c) {
    return __builtin_amdgcn_wmma_f32_16x16x32_bf16(
        /*neg_a=*/false, a, /*neg_b=*/false, b,
        /*c_mod=*/(short)0, c, /*reuse_a=*/false, /*reuse_b=*/false);
}

// ---------------------------------------------------------------------------
// f32 -> bf16 elementwise convert
// ---------------------------------------------------------------------------
__global__ __launch_bounds__(256) void cvt_f32_bf16(const float* __restrict__ in,
                                                    bf16_t* __restrict__ out, int n) {
    int i = blockIdx.x * 256 + threadIdx.x;
    if (i < n) out[i] = (bf16_t)in[i];
}

// ---------------------------------------------------------------------------
// f32 [rows,cols] -> bf16 [cols,rows] tiled transpose through LDS
// ---------------------------------------------------------------------------
__global__ __launch_bounds__(256) void transpose_cvt(const float* __restrict__ in,
                                                     bf16_t* __restrict__ out,
                                                     int rows, int cols) {
    __shared__ bf16_t tile[32][33];
    const int c0 = blockIdx.x * 32, r0 = blockIdx.y * 32;
    const int tx = threadIdx.x & 31, ty = threadIdx.x >> 5;
    #pragma unroll
    for (int rr = ty; rr < 32; rr += 8) {
        int r = r0 + rr, c = c0 + tx;
        tile[rr][tx] = (r < rows && c < cols) ? (bf16_t)in[(size_t)r * cols + c]
                                              : (bf16_t)0.0f;
    }
    __syncthreads();
    #pragma unroll
    for (int cc = ty; cc < 32; cc += 8) {
        int c = c0 + cc, r = r0 + tx;
        if (c < cols && r < rows) out[(size_t)c * rows + r] = tile[tx][cc];
    }
}

// ---------------------------------------------------------------------------
// QKV GEMM: [8192,1024]bf16 @ [1024,3072] (WT stored [3072,1024]) -> scatter
//   q -> Qb[b*16+h][n][d] * SCALE   (bf16)
//   k -> Kb[b*16+h][n][d]           (bf16)
//   v -> VTb[b*16+h][d][n]          (bf16, transposed for PV B-fragments)
// Block: 256 thr = 8 waves; block tile 128x128; wave tile 64x32 (4x2 WMMA).
// ---------------------------------------------------------------------------
__global__ __launch_bounds__(256) void qkv_gemm(const bf16_t* __restrict__ X,
                                                const bf16_t* __restrict__ WT,
                                                bf16_t* __restrict__ Qb,
                                                bf16_t* __restrict__ Kb,
                                                bf16_t* __restrict__ VTb) {
    const int lane = threadIdx.x & 31;
    const int wave = threadIdx.x >> 5;
    const int m0 = blockIdx.x * 128 + (wave & 1) * 64;
    const int n0 = blockIdx.y * 128 + (wave >> 1) * 32;

    v8f acc[4][2] = {};
    for (int k0 = 0; k0 < 1024; k0 += 32) {
        v16bf a[4], b[2];
        #pragma unroll
        for (int i = 0; i < 4; ++i) a[i] = load_a_frag(X, 1024, m0 + 16 * i, k0, lane);
        #pragma unroll
        for (int j = 0; j < 2; ++j) b[j] = load_b_frag(WT, 1024, n0 + 16 * j, k0, lane);
        #pragma unroll
        for (int i = 0; i < 4; ++i)
            #pragma unroll
            for (int j = 0; j < 2; ++j)
                acc[i][j] = wmma_bf16(a[i], b[j], acc[i][j]);
    }

    const int half = lane >> 4, nl = lane & 15;
    #pragma unroll
    for (int i = 0; i < 4; ++i)
        #pragma unroll
        for (int j = 0; j < 2; ++j)
            #pragma unroll
            for (int r = 0; r < 8; ++r) {
                int row = m0 + 16 * i + r + 8 * half;   // 0..8191 = b*1024+nn
                int col = n0 + 16 * j + nl;             // 0..3071
                float v = acc[i][j][r];
                int bb = row >> 10, nn = row & 1023;
                int sec = col >> 10, cc = col & 1023;
                int h = cc >> 6, d = cc & 63;
                size_t bh = (size_t)(bb * 16 + h);
                if (sec == 0)
                    Qb[(bh * 1024 + nn) * 64 + d] = (bf16_t)(v * SCALE_F);
                else if (sec == 1)
                    Kb[(bh * 1024 + nn) * 64 + d] = (bf16_t)v;
                else
                    VTb[(bh * 64 + d) * 1024 + nn] = (bf16_t)v;
            }
}

// ---------------------------------------------------------------------------
// Flash attention: O = softmax(Q Kᵀ) V per (b,h).  Q pre-scaled.
// Block: 128 thr = 4 waves; each wave owns 16 Q rows; block = 64 rows.
// Key blocks of 64: 8 WMMA for S, online softmax, P via LDS transpose,
// 8 WMMA for PV.  Writes attn_ret (f32 [b,h,n,d]) + bf16 merged copy.
// ---------------------------------------------------------------------------
__global__ __launch_bounds__(128) void flash_attn(const bf16_t* __restrict__ Qb,
                                                  const bf16_t* __restrict__ Kb,
                                                  const bf16_t* __restrict__ VTb,
                                                  float* __restrict__ Oattn,
                                                  bf16_t* __restrict__ Om) {
    const int lane = threadIdx.x & 31;
    const int wave = threadIdx.x >> 5;
    const int bh = blockIdx.x;            // 0..127
    const int bb = bh >> 4, h = bh & 15;
    const int r0 = blockIdx.y * 64 + wave * 16;

    const bf16_t* Qp = Qb  + (size_t)bh * 1024 * 64;
    const bf16_t* Kp = Kb  + (size_t)bh * 1024 * 64;
    const bf16_t* Vp = VTb + (size_t)bh * 64 * 1024;

    __shared__ __align__(16) bf16_t pshare[4][16][72];   // per-wave 16x64 (padded)

    v16bf qa0 = load_a_frag(Qp, 64, r0, 0, lane);
    v16bf qa1 = load_a_frag(Qp, 64, r0, 32, lane);

    float mrow[8], lrow[8];
    #pragma unroll
    for (int r = 0; r < 8; ++r) { mrow[r] = -1e30f; lrow[r] = 0.0f; }
    v8f oacc[4] = {};

    const int half = lane >> 4, nl = lane & 15;

    for (int j0 = 0; j0 < 1024; j0 += 64) {
        // ---- S = Q Kᵀ for 16 rows x 64 keys ----
        v8f s[4] = {};
        #pragma unroll
        for (int t = 0; t < 4; ++t) {
            v16bf bk0 = load_b_frag(Kp, 64, j0 + 16 * t, 0, lane);
            v16bf bk1 = load_b_frag(Kp, 64, j0 + 16 * t, 32, lane);
            s[t] = wmma_bf16(qa0, bk0, s[t]);
            s[t] = wmma_bf16(qa1, bk1, s[t]);
        }
        // ---- online softmax: rows live across 16 lanes of one acc slot ----
        #pragma unroll
        for (int r = 0; r < 8; ++r) {
            float mb = fmaxf(fmaxf(s[0][r], s[1][r]), fmaxf(s[2][r], s[3][r]));
            #pragma unroll
            for (int off = 8; off >= 1; off >>= 1)
                mb = fmaxf(mb, __shfl_xor(mb, off, 32));
            float mnew  = fmaxf(mrow[r], mb);
            float alpha = __expf(mrow[r] - mnew);
            float psum = 0.0f;
            #pragma unroll
            for (int t = 0; t < 4; ++t) {
                float p = __expf(s[t][r] - mnew);
                s[t][r] = p;
                psum += p;
            }
            #pragma unroll
            for (int off = 8; off >= 1; off >>= 1)
                psum += __shfl_xor(psum, off, 32);
            lrow[r] = lrow[r] * alpha + psum;
            mrow[r] = mnew;
            oacc[0][r] *= alpha; oacc[1][r] *= alpha;
            oacc[2][r] *= alpha; oacc[3][r] *= alpha;
        }
        // ---- C-layout P -> LDS (row-major 16x64) -> A-layout fragments ----
        #pragma unroll
        for (int t = 0; t < 4; ++t)
            #pragma unroll
            for (int r = 0; r < 8; ++r)
                pshare[wave][r + 8 * half][16 * t + nl] = (bf16_t)s[t][r];
        // same-wave LDS ops are in-order; compiler inserts s_wait_dscnt
        v16bf pa0 = load_a_frag(&pshare[wave][0][0], 72, 0, 0,  lane);
        v16bf pa1 = load_a_frag(&pshare[wave][0][0], 72, 0, 32, lane);
        // ---- O += P V  (V stored transposed: contiguous along keys) ----
        #pragma unroll
        for (int t = 0; t < 4; ++t) {
            v16bf bv0 = load_b_frag(Vp, 1024, 16 * t, j0,      lane);
            v16bf bv1 = load_b_frag(Vp, 1024, 16 * t, j0 + 32, lane);
            oacc[t] = wmma_bf16(pa0, bv0, oacc[t]);
            oacc[t] = wmma_bf16(pa1, bv1, oacc[t]);
        }
    }
    // ---- epilogue: normalize, write attn_ret f32 + merged bf16 ----
    #pragma unroll
    for (int r = 0; r < 8; ++r) {
        float inv = 1.0f / lrow[r];
        int row = r0 + r + 8 * half;
        #pragma unroll
        for (int t = 0; t < 4; ++t) {
            int d = 16 * t + nl;
            float o = oacc[t][r] * inv;
            Oattn[((size_t)bh * 1024 + row) * 64 + d] = o;
            Om[((size_t)(bb * 1024 + row)) * 1024 + h * 64 + d] = (bf16_t)o;
        }
    }
}

// ---------------------------------------------------------------------------
// Projection: proj = Om[8192,1024] @ W_out + b_out  (WoutT stored [1024,1024])
// ---------------------------------------------------------------------------
__global__ __launch_bounds__(256) void proj_gemm(const bf16_t* __restrict__ A,
                                                 const bf16_t* __restrict__ WT,
                                                 const float* __restrict__ bias,
                                                 float* __restrict__ C) {
    const int lane = threadIdx.x & 31;
    const int wave = threadIdx.x >> 5;
    const int m0 = blockIdx.x * 128 + (wave & 1) * 64;
    const int n0 = blockIdx.y * 128 + (wave >> 1) * 32;

    v8f acc[4][2] = {};
    for (int k0 = 0; k0 < 1024; k0 += 32) {
        v16bf a[4], b[2];
        #pragma unroll
        for (int i = 0; i < 4; ++i) a[i] = load_a_frag(A, 1024, m0 + 16 * i, k0, lane);
        #pragma unroll
        for (int j = 0; j < 2; ++j) b[j] = load_b_frag(WT, 1024, n0 + 16 * j, k0, lane);
        #pragma unroll
        for (int i = 0; i < 4; ++i)
            #pragma unroll
            for (int j = 0; j < 2; ++j)
                acc[i][j] = wmma_bf16(a[i], b[j], acc[i][j]);
    }
    const int half = lane >> 4, nl = lane & 15;
    #pragma unroll
    for (int i = 0; i < 4; ++i)
        #pragma unroll
        for (int j = 0; j < 2; ++j)
            #pragma unroll
            for (int r = 0; r < 8; ++r) {
                int row = m0 + 16 * i + r + 8 * half;
                int col = n0 + 16 * j + nl;
                C[(size_t)row * 1024 + col] = acc[i][j][r] + bias[col];
            }
}

// ---------------------------------------------------------------------------
// Orchestration
// ---------------------------------------------------------------------------
extern "C" void kernel_launch(void* const* d_in, const int* in_sizes, int n_in,
                              void* d_out, int out_size, void* d_ws, size_t ws_size,
                              hipStream_t stream) {
    (void)in_sizes; (void)n_in; (void)out_size; (void)ws_size;
    const float* x    = (const float*)d_in[0];   // [8,1024,1024]
    const float* Wqkv = (const float*)d_in[1];   // [1024,3072]
    const float* Wout = (const float*)d_in[2];   // [1024,1024]
    const float* bout = (const float*)d_in[3];   // [1024]

    float* proj = (float*)d_out;                        // [8,1024,1024]
    float* attn = proj + (size_t)8 * 1024 * 1024;       // [8,16,1024,64]

    bf16_t* Xb    = (bf16_t*)d_ws;                      // 8192*1024
    bf16_t* WqkvT = Xb    + (size_t)8192 * 1024;        // 3072*1024
    bf16_t* WoutT = WqkvT + (size_t)3072 * 1024;        // 1024*1024
    bf16_t* Qb    = WoutT + (size_t)1024 * 1024;        // 128*1024*64
    bf16_t* Kb    = Qb    + (size_t)128 * 1024 * 64;
    bf16_t* VTb   = Kb    + (size_t)128 * 1024 * 64;
    bf16_t* Omb   = VTb   + (size_t)128 * 1024 * 64;    // 8192*1024

    cvt_f32_bf16<<<(8192 * 1024) / 256, 256, 0, stream>>>(x, Xb, 8192 * 1024);
    transpose_cvt<<<dim3(96, 32), 256, 0, stream>>>(Wqkv, WqkvT, 1024, 3072);
    transpose_cvt<<<dim3(32, 32), 256, 0, stream>>>(Wout, WoutT, 1024, 1024);

    qkv_gemm<<<dim3(64, 24), 256, 0, stream>>>(Xb, WqkvT, Qb, Kb, VTb);
    flash_attn<<<dim3(128, 16), 128, 0, stream>>>(Qb, Kb, VTb, attn, Omb);
    proj_gemm<<<dim3(64, 8), 256, 0, stream>>>(Omb, WoutT, bout, proj);
}
